// DecoupledSOLOHead_60876866453719
// MI455X (gfx1250) — 8-bit matrix operand order repaired and checked by reading, compiled, not run on hardware
//
#include <hip/hip_runtime.h>
#include <hip/hip_bf16.h>

#define MASK_THR 0.005f
#define SIGMA_K  2.0f
#define HH 200
#define WW 304
#define HW (HH * WW)

#define KSPLIT  5
#define KSEG    (HW / KSPLIT)
#define KSTAGE  128
#define NSTAGES (KSEG / KSTAGE)
#define LROW    144

typedef __attribute__((ext_vector_type(2))) unsigned int v2u;
typedef __attribute__((ext_vector_type(4))) unsigned int v4u;
typedef __attribute__((ext_vector_type(8))) int          v8i;

__global__ void build_masks(const float* __restrict__ segx,
                            const float* __restrict__ segy,
                            const float* __restrict__ cate_scores,
                            const int*   __restrict__ x_inds,
                            const int*   __restrict__ y_inds,
                            unsigned char* __restrict__ mask,
                            float* __restrict__ sum_masks,
                            float* __restrict__ scores,
                            int N)
{
    const int n   = blockIdx.x;
    const int tid = threadIdx.x;
    unsigned char* mrow = mask + (size_t)n * HW;

    __shared__ float sf[256];
    __shared__ int   si[256];

    if (n >= N) {
        for (int i = tid * 4; i < HW; i += 256 * 4)
            *(unsigned int*)(mrow + i) = 0u;
        if (tid == 0) { sum_masks[n] = 0.0f; scores[n] = 0.0f; }
        return;
    }

    const float* xr = segx + (size_t)x_inds[n] * HW;
    const float* yr = segy + (size_t)y_inds[n] * HW;

    int   cnt  = 0;
    float ssum = 0.0f;
    for (int i = tid * 4; i < HW; i += 256 * 4) {
        const float4 xv = *(const float4*)(xr + i);
        const float4 yv = *(const float4*)(yr + i);
        const float s0 = xv.x * yv.x;
        const float s1 = xv.y * yv.y;
        const float s2 = xv.z * yv.z;
        const float s3 = xv.w * yv.w;
        unsigned int p = 0u;
        if (s0 > MASK_THR) { p |= 1u;       cnt++; ssum += s0; }
        if (s1 > MASK_THR) { p |= 1u << 8;  cnt++; ssum += s1; }
        if (s2 > MASK_THR) { p |= 1u << 16; cnt++; ssum += s2; }
        if (s3 > MASK_THR) { p |= 1u << 24; cnt++; ssum += s3; }
        *(unsigned int*)(mrow + i) = p;
    }

    sf[tid] = ssum; si[tid] = cnt;
    __syncthreads();
    for (int off = 128; off > 0; off >>= 1) {
        if (tid < off) { sf[tid] += sf[tid + off]; si[tid] += si[tid + off]; }
        __syncthreads();
    }
    if (tid == 0) {
        const float area = (float)si[0];
        sum_masks[n] = area;
        scores[n] = cate_scores[n] * (sf[0] / fmaxf(area, 1.0f));
    }
}

__device__ __forceinline__ void issue4(const unsigned char* base,
                                       const unsigned int* goff,
                                       const unsigned int* loff,
                                       unsigned int k)
{
#pragma unroll
    for (int q = 0; q < 4; ++q)
        asm volatile("global_load_async_to_lds_b128 %0, %1, %2"
                     :: "v"(loff[q]), "v"(goff[q] + k), "s"(base)
                     : "memory");
}

__global__ void __launch_bounds__(128)
mask_gemm(const unsigned char* __restrict__ mask,
          int* __restrict__ interPart, int NP)
{
    __shared__ __align__(16) unsigned char Abuf[2][32][LROW];
    __shared__ __align__(16) unsigned char Bbuf[2][32][LROW];

    const int tid  = threadIdx.x;
    const int lane = tid & 31;
    const int wave = tid >> 5;
    const int rowBase = blockIdx.y * 32;
    const int colBase = blockIdx.x * 32;
    const unsigned int kBase0 = blockIdx.z * KSEG;

    unsigned int goff[4], loff0[4], loff1[4];
#pragma unroll
    for (int q = 0; q < 4; ++q) {
        const int c   = q * 128 + tid;
        const int isB = c >> 8;
        const int cc  = c & 255;
        const int row = cc >> 3;
        const int ch  = cc & 7;
        const int grow = (isB ? colBase : rowBase) + row;
        goff[q]  = (unsigned int)((size_t)grow * HW) + kBase0 + ch * 16;
        loff0[q] = (unsigned int)(unsigned long long)
                   ((isB ? &Bbuf[0][row][0] : &Abuf[0][row][0]) + ch * 16);
        loff1[q] = (unsigned int)(unsigned long long)
                   ((isB ? &Bbuf[1][row][0] : &Abuf[1][row][0]) + ch * 16);
    }

    const int arow  = (wave >> 1) * 16 + (lane & 15);
    const int abase = (lane >> 4) * 16;
    const int brow  = (wave & 1) * 16 + (lane & 15);
    const int bbase = (lane >> 4) * 16;

    issue4(mask, goff, loff0, 0u);

    v8i c = {};
    for (int s = 0; s < NSTAGES; ++s) {
        const int buf = s & 1;
        if (s + 1 < NSTAGES) {
            issue4(mask, goff, (((s + 1) & 1) ? loff1 : loff0),
                   (unsigned int)(s + 1) * KSTAGE);
            asm volatile("s_wait_asynccnt 4" ::: "memory");
        } else {
            asm volatile("s_wait_asynccnt 0" ::: "memory");
        }
        __syncthreads();

        const unsigned char* arowp = &Abuf[buf][arow][0];
        const unsigned char* browp = &Bbuf[buf][brow][0];
#pragma unroll
        for (int kk = 0; kk < KSTAGE; kk += 64) {
            const unsigned char* ap = arowp + kk + abase;
            const unsigned char* bp = browp + kk + bbase;
            const v2u a01 = *(const v2u*)(ap);
            const v2u a23 = *(const v2u*)(ap + 8);
            const v2u a45 = *(const v2u*)(ap + 32);
            const v2u a67 = *(const v2u*)(ap + 40);
            const v4u b03 = *(const v4u*)(bp);
            const v4u b47 = *(const v4u*)(bp + 32);
            v8i A, B;
            A[0] = (int)a01.x; A[1] = (int)a01.y;
            A[2] = (int)a23.x; A[3] = (int)a23.y;
            A[4] = (int)a45.x; A[5] = (int)a45.y;
            A[6] = (int)a67.x; A[7] = (int)a67.y;
            B[0] = (int)b03.x; B[1] = (int)b03.y;
            B[2] = (int)b03.z; B[3] = (int)b03.w;
            B[4] = (int)b47.x; B[5] = (int)b47.y;
            B[6] = (int)b47.z; B[7] = (int)b47.w;
            c = __builtin_amdgcn_wmma_i32_16x16x64_iu8(false, A, false, B, c,
                                                       false, false);
        }
        __syncthreads();
    }

    int* outp = interPart + (size_t)blockIdx.z * NP * NP;
    const int orow = rowBase + (wave >> 1) * 16 + ((lane >> 4) ? 8 : 0);
    const int ocol = colBase + (wave & 1) * 16 + (lane & 15);
#pragma unroll
    for (int r = 0; r < 8; ++r)
        outp[(size_t)(orow + r) * NP + ocol] = c[r];
}

__global__ void reduce_parts(const int* __restrict__ part,
                             int* __restrict__ inter, int nElem)
{
    const int idx = blockIdx.x * 256 + threadIdx.x;
    if (idx < nElem) {
        int s = 0;
#pragma unroll
        for (int p = 0; p < KSPLIT; ++p)
            s += part[(size_t)p * nElem + idx];
        inter[idx] = s;
    }
}

__global__ void comp_max(const int* __restrict__ inter,
                         const float* __restrict__ sum_masks,
                         const int* __restrict__ labels,
                         float* __restrict__ comp, int N, int NP)
{
    const int j   = blockIdx.x;
    const int tid = threadIdx.x;
    __shared__ float sm[256];

    const int   lj = labels[j];
    const float sj = sum_masks[j];
    float m = 0.0f;
    for (int i = tid; i < j; i += 256) {
        if (labels[i] == lj) {
            const float in = (float)inter[(size_t)i * NP + j];
            const float un = sum_masks[i] + sj - in;
            m = fmaxf(m, in / fmaxf(un, 1e-6f));
        }
    }
    sm[tid] = m;
    __syncthreads();
    for (int off = 128; off > 0; off >>= 1) {
        if (tid < off) sm[tid] = fmaxf(sm[tid], sm[tid + off]);
        __syncthreads();
    }
    if (tid == 0) comp[j] = sm[0];
}

__global__ void decay_min(const int* __restrict__ inter,
                          const float* __restrict__ sum_masks,
                          const int* __restrict__ labels,
                          const float* __restrict__ comp,
                          const float* __restrict__ scores,
                          float* __restrict__ out, int N, int NP)
{
    const int j   = blockIdx.x;
    const int tid = threadIdx.x;
    __shared__ float sm[256];

    const int   lj = labels[j];
    const float sj = sum_masks[j];
    float mn = 3.402823466e+38f;
    for (int i = tid; i < N; i += 256) {
        float diou = 0.0f;
        if (i < j && labels[i] == lj) {
            const float in = (float)inter[(size_t)i * NP + j];
            const float un = sum_masks[i] + sj - in;
            diou = in / fmaxf(un, 1e-6f);
        }
        const float ci    = comp[i];
        const float ratio = __expf(SIGMA_K * (ci * ci - diou * diou));
        mn = fminf(mn, ratio);
    }
    sm[tid] = mn;
    __syncthreads();
    for (int off = 128; off > 0; off >>= 1) {
        if (tid < off) sm[tid] = fminf(sm[tid], sm[tid + off]);
        __syncthreads();
    }
    if (tid == 0) out[j] = scores[j] * sm[0];
}

extern "C" void kernel_launch(void* const* d_in, const int* in_sizes, int n_in,
                              void* d_out, int out_size, void* d_ws, size_t ws_size,
                              hipStream_t stream)
{
    const float* cate_scores = (const float*)d_in[0];
    const float* segx        = (const float*)d_in[1];
    const float* segy        = (const float*)d_in[2];
    const int*   labels      = (const int*)d_in[3];
    const int*   x_inds      = (const int*)d_in[4];
    const int*   y_inds      = (const int*)d_in[5];
    float*       out         = (float*)d_out;

    const int N  = in_sizes[0];
    const int NP = (N + 31) & ~31;

    char* ws = (char*)d_ws;
    size_t off = 0;
    unsigned char* mask = (unsigned char*)(ws + off);
    off += ((size_t)NP * HW + 255) & ~(size_t)255;
    int* interPart = (int*)(ws + off);
    off += ((size_t)KSPLIT * NP * NP * sizeof(int) + 255) & ~(size_t)255;
    int* inter = (int*)(ws + off);
    off += ((size_t)NP * NP * sizeof(int) + 255) & ~(size_t)255;
    float* sum_masks = (float*)(ws + off); off += (size_t)NP * sizeof(float);
    float* scores    = (float*)(ws + off); off += (size_t)NP * sizeof(float);
    float* comp      = (float*)(ws + off); off += (size_t)NP * sizeof(float);
    (void)off; (void)ws_size; (void)n_in; (void)out_size;

    build_masks<<<NP, 256, 0, stream>>>(segx, segy, cate_scores, x_inds, y_inds,
                                        mask, sum_masks, scores, N);

    dim3 gemm_grid(NP / 32, NP / 32, KSPLIT);
    mask_gemm<<<gemm_grid, 128, 0, stream>>>(mask, interPart, NP);

    const int nElem = NP * NP;
    reduce_parts<<<(nElem + 255) / 256, 256, 0, stream>>>(interPart, inter, nElem);

    comp_max<<<N, 256, 0, stream>>>(inter, sum_masks, labels, comp, N, NP);
    decay_min<<<N, 256, 0, stream>>>(inter, sum_masks, labels, comp, scores,
                                     out, N, NP);
}
